// BottleneckBlock_69930657513782
// MI455X (gfx1250) — compile-verified, matrix operand order checked by reading, not run on hardware
//
#include <hip/hip_runtime.h>
#include <hip/hip_bf16.h>
#include <math.h>

typedef __attribute__((ext_vector_type(16))) __bf16   v16bf;
typedef __attribute__((ext_vector_type(8)))  float    v8f;
typedef __attribute__((ext_vector_type(4)))  unsigned uint4v;

typedef __attribute__((ext_vector_type(4)))  unsigned tdm_g0_t;
typedef __attribute__((ext_vector_type(8)))  int      tdm_g1_t;
typedef __attribute__((ext_vector_type(4)))  int      tdm_g2_t;
typedef __attribute__((ext_vector_type(8)))  int      tdm_g4_t;

#define NB     32
#define WIDTH  512
#define TT     2048
#define KBINS  2048
#define ROWS   (NB * TT)          /* 65536 rows of xf */
#define NELEM  (ROWS * WIDTH)     /* 33554432 elements of x */
#define JBLK   (KBINS / 16)       /* 128 column blocks */
#define XSTR   (WIDTH + 8)        /* LDS row stride (ushorts); 1040B = 65*16B, 4-bank skew */
#define KTWORDS (16 * XSTR)       /* one k-tile buffer: 8320 ushorts = 16640 B */

#define STATS_BLOCKS 1024
#define BV_BLOCKS    256

__device__ __forceinline__ unsigned short f2bf(float f) {
    unsigned u = __float_as_uint(f);
    u += 0x7FFFu + ((u >> 16) & 1u);      // round-to-nearest-even
    return (unsigned short)(u >> 16);
}

// ---------------------------------------------------------------------------
// TDM: DMA one 16-row x 512-col bf16 codebook tile (row = 1KB) into LDS,
// inserting 16B of padding after every 1KB row (=> LDS row stride 1040B).
// D# packing per CDNA5 ISA ch.8 (2D tile, 4-byte units).
// ---------------------------------------------------------------------------
__device__ __forceinline__ void tdm_load_ktile(const unsigned short* kbf,
                                               int jb, unsigned lds_addr) {
    unsigned long long ga = (unsigned long long)(size_t)kbf
                          + (unsigned long long)jb * (16u * 1024u); // 16 rows * 1KB
    tdm_g0_t g0;
    g0.x = 1u;                                            // count=1, user descriptor
    g0.y = lds_addr;                                      // LDS byte address
    g0.z = (unsigned)(ga & 0xFFFFFFFFull);                // global_addr[31:0]
    g0.w = ((unsigned)(ga >> 32) & 0x01FFFFFFu)           // global_addr[56:32]
         | 0x80000000u;                                   // type=2 ("image")
    tdm_g1_t g1;
    g1[0] = (2 << 16)        // data_size = 4B units
          | (1 << 20)        // pad_enable
          | (7 << 22)        // pad_interval: 256 DWORDs (= one 1KB row)
          | (3 << 25);       // pad_amount: 4 DWORDs (16B)
    g1[1] = 256 << 16;       // tensor_dim0 = 256 DW per row (bits 79:48)
    g1[2] = 2048 << 16;      // tensor_dim1 = 2048 rows (bits 111:80)
    g1[3] = 256 << 16;       // tile_dim0 = 256 DW (bits 127:112)
    g1[4] = 16;              // tile_dim1 = 16 rows
    g1[5] = 256;             // tensor_dim0_stride = 256 DW
    g1[6] = 0;
    g1[7] = 0;
    tdm_g2_t gz = {0, 0, 0, 0};
    tdm_g4_t g4 = {0, 0, 0, 0, 0, 0, 0, 0};
    __builtin_amdgcn_tensor_load_to_lds(g0, g1, gz, gz, g4, 0);
}

// ---------------------------------------------------------------------------
// Convert codebook to bf16 and compute ||k_j||^2
// ---------------------------------------------------------------------------
__global__ __launch_bounds__(64)
void prep_k_kernel(const float* __restrict__ k,
                   unsigned short* __restrict__ kbf,
                   float* __restrict__ knorm) {
    __shared__ float red[64];
    const int j   = blockIdx.x;
    const int tid = threadIdx.x;
    const float* krow = k + (size_t)j * WIDTH;
    unsigned short* brow = kbf + (size_t)j * WIDTH;
    float s = 0.f;
    for (int c = tid; c < WIDTH; c += 64) {
        float v = krow[c];
        s += v * v;
        brow[c] = f2bf(v);
    }
    red[tid] = s;
    __syncthreads();
    for (int st = 32; st > 0; st >>= 1) {
        if (tid < st) red[tid] += red[tid + st];
        __syncthreads();
    }
    if (tid == 0) knorm[j] = red[0];
}

// ---------------------------------------------------------------------------
// Per-block partial sums of x and x^2 (deterministic two-pass reduction)
// ---------------------------------------------------------------------------
__global__ __launch_bounds__(256)
void stats_kernel(const float* __restrict__ x,
                  float* __restrict__ psx, float* __restrict__ psxx) {
    __shared__ float rs[256], rss[256];
    float sx = 0.f, sxx = 0.f;
    const size_t stride = (size_t)gridDim.x * blockDim.x;
    for (size_t i = (size_t)blockIdx.x * blockDim.x + threadIdx.x; i < (size_t)NELEM; i += stride) {
        float v = x[i];
        sx += v; sxx += v * v;
    }
    rs[threadIdx.x] = sx; rss[threadIdx.x] = sxx;
    __syncthreads();
    for (int st = 128; st > 0; st >>= 1) {
        if (threadIdx.x < st) {
            rs[threadIdx.x]  += rs[threadIdx.x + st];
            rss[threadIdx.x] += rss[threadIdx.x + st];
        }
        __syncthreads();
    }
    if (threadIdx.x == 0) { psx[blockIdx.x] = rs[0]; psxx[blockIdx.x] = rss[0]; }
}

// ---------------------------------------------------------------------------
// Main fused kernel: bf16 WMMA GEMM (xf * k^T) + per-row argmin.
//  - A fragments (16 rows x K=512) hoisted into 128 VGPRs/lane (CDNA5 has 1024)
//  - k tiles double-buffered in LDS, streamed by the Tensor Data Mover with
//    hardware row padding; synchronized via TENSORcnt + workgroup barrier.
// ---------------------------------------------------------------------------
__global__ __launch_bounds__(64)
void argmin_gemm_kernel(const float* __restrict__ x,
                        const unsigned short* __restrict__ kbf,
                        const float* __restrict__ knorm,
                        float* __restrict__ best_val,
                        int*   __restrict__ best_idx) {
    // Phase 1: x tile (32 rows x 512 K).  Phase 2: two 16x512 k-tile buffers.
    __shared__ unsigned short smem[32 * XSTR];   // 33,280 B

    const int tid  = threadIdx.x;     // 0..63
    const int wave = tid >> 5;        // 0..1
    const int lane = tid & 31;
    const int R0   = blockIdx.x * 32; // global row base (rows = n*T + t)
    const int n    = R0 >> 11;        // 32 rows never straddle n
    const int t0   = R0 & 2047;

    // ---- stage x tile transposed: smem[t_local][c] = bf16(x[n][c][t0+t_local])
    for (int q = tid; q < WIDTH * 8; q += 64) {
        const int c   = q >> 3;
        const int tl4 = (q & 7) << 2;
        const float4 v = *(const float4*)(x + (((size_t)n * WIDTH + c) * TT + t0 + tl4));
        smem[(tl4 + 0) * XSTR + c] = f2bf(v.x);
        smem[(tl4 + 1) * XSTR + c] = f2bf(v.y);
        smem[(tl4 + 2) * XSTR + c] = f2bf(v.z);
        smem[(tl4 + 3) * XSTR + c] = f2bf(v.w);
    }
    __syncthreads();

    const int half = lane >> 4;       // K-half select (16-bit A/B layouts)
    const int l15  = lane & 15;       // A: row M; B: col N; C/D: col N
    const int mrow = wave * 16;       // this wave's 16-row slab

    // ---- hoist all A fragments for K=512 into registers (16 x 8 VGPRs)
    union Frag { uint4v u[2]; v16bf b; };
    Frag afrag[16];
#pragma unroll
    for (int kb = 0; kb < 16; ++kb) {
        const unsigned short* ap = smem + (mrow + l15) * XSTR + kb * 32 + half * 8;
        afrag[kb].u[0] = *(const uint4v*)(ap);
        afrag[kb].u[1] = *(const uint4v*)(ap + 16);
    }
    __syncthreads();   // barrier emits s_wait_dscnt 0 -> x tile fully read; LDS reusable

    // low 32 bits of the flat shared address == workgroup-relative LDS offset
    const unsigned kt0_addr = (unsigned)(size_t)(&smem[0]);
    const unsigned kt1_addr = (unsigned)(size_t)(&smem[KTWORDS]);

    if (wave == 0) {
        tdm_load_ktile(kbf, 0, kt0_addr);
        __builtin_amdgcn_s_wait_tensorcnt(0);
    }
    __syncthreads();   // buffer 0 valid for the whole workgroup

    float minv[8];
    int   mini[8];
#pragma unroll
    for (int r = 0; r < 8; ++r) { minv[r] = 3.4e38f; mini[r] = 0; }

    int cur = 0;
    for (int jb = 0; jb < JBLK; ++jb) {
        // prefetch next k tile into the other buffer while we compute
        if (wave == 0 && jb + 1 < JBLK)
            tdm_load_ktile(kbf, jb + 1, cur ? kt0_addr : kt1_addr);

        const unsigned short* ktp = smem + (cur ? KTWORDS : 0);
        v8f acc = {};
#pragma unroll
        for (int kb = 0; kb < 16; ++kb) {
            Frag fb;
            const unsigned short* bp = ktp + l15 * XSTR + kb * 32 + half * 8;
            fb.u[0] = *(const uint4v*)(bp);
            fb.u[1] = *(const uint4v*)(bp + 16);
            acc = __builtin_amdgcn_wmma_f32_16x16x32_bf16(
                false, afrag[kb].b, false, fb.b, (short)0, acc, false, false);
        }

        const int   j  = jb * 16 + l15;
        const float kn = knorm[j];
#pragma unroll
        for (int r = 0; r < 8; ++r) {
            const float v = fmaf(-2.0f, acc[r], kn);  // ||k_j||^2 - 2<x,k_j>
            if (v < minv[r]) { minv[r] = v; mini[r] = j; }
        }

        __syncthreads();                               // all waves done reading cur
        if (wave == 0) __builtin_amdgcn_s_wait_tensorcnt(0);
        __syncthreads();                               // next buffer valid
        cur ^= 1;
    }

    // argmin across the 16 lanes sharing each C-row (xor masks preserve lane[4])
#pragma unroll
    for (int r = 0; r < 8; ++r) {
        float v = minv[r];
        int   i = mini[r];
#pragma unroll
        for (int s = 8; s > 0; s >>= 1) {
            const float ov = __shfl_xor(v, s, 32);
            const int   oi = __shfl_xor(i, s, 32);
            if (ov < v || (ov == v && oi < i)) { v = ov; i = oi; }
        }
        if (l15 == 0) {
            const int row = R0 + wave * 16 + half * 8 + r;  // C layout: M = r + 8*lane[4]
            best_val[row] = v;
            best_idx[row] = i;
        }
    }
}

// ---------------------------------------------------------------------------
// Per-block partial sums of best_val (deterministic)
// ---------------------------------------------------------------------------
__global__ __launch_bounds__(256)
void reduce_bv_kernel(const float* __restrict__ best_val, float* __restrict__ pbv) {
    __shared__ float red[256];
    float s = 0.f;
    for (int i = blockIdx.x * blockDim.x + threadIdx.x; i < ROWS; i += gridDim.x * blockDim.x)
        s += best_val[i];
    red[threadIdx.x] = s;
    __syncthreads();
    for (int st = 128; st > 0; st >>= 1) {
        if (threadIdx.x < st) red[threadIdx.x] += red[threadIdx.x + st];
        __syncthreads();
    }
    if (threadIdx.x == 0) pbv[blockIdx.x] = red[0];
}

// ---------------------------------------------------------------------------
// Single-block finalize: commit_loss, fit, prenorm
// ---------------------------------------------------------------------------
__global__ __launch_bounds__(256)
void finalize_kernel(const float* __restrict__ psx, const float* __restrict__ psxx,
                     const float* __restrict__ pbv, float* __restrict__ out_tail) {
    __shared__ float a[256], b[256], c[256];
    float sx = 0.f, sxx = 0.f, bv = 0.f;
    for (int i = threadIdx.x; i < STATS_BLOCKS; i += 256) { sx += psx[i]; sxx += psxx[i]; }
    for (int i = threadIdx.x; i < BV_BLOCKS;    i += 256) { bv += pbv[i]; }
    a[threadIdx.x] = sx; b[threadIdx.x] = sxx; c[threadIdx.x] = bv;
    __syncthreads();
    for (int st = 128; st > 0; st >>= 1) {
        if (threadIdx.x < st) {
            a[threadIdx.x] += a[threadIdx.x + st];
            b[threadIdx.x] += b[threadIdx.x + st];
            c[threadIdx.x] += c[threadIdx.x + st];
        }
        __syncthreads();
    }
    if (threadIdx.x == 0) {
        const float n_elem = (float)NELEM;
        const float sumx = a[0], sumxx = b[0], bsum = c[0];
        out_tail[0] = (bsum + sumxx) / n_elem;                    // commit_loss
        out_tail[1] = (bsum + sumxx) / (float)ROWS;               // fit
        const float mean = sumx / n_elem;
        out_tail[2] = sqrtf(fmaxf(0.f, sumxx - n_elem * mean * mean) / n_elem); // prenorm
    }
}

// ---------------------------------------------------------------------------
// x_l (as float) and x_d gather (straight-through => exactly k[x_l])
// ---------------------------------------------------------------------------
__global__ __launch_bounds__(256)
void write_xl_kernel(const int* __restrict__ best_idx, float* __restrict__ out_xl) {
    const int i = blockIdx.x * blockDim.x + threadIdx.x;
    if (i < ROWS) out_xl[i] = (float)best_idx[i];
}

__global__ __launch_bounds__(256)
void gather_kernel(const float* __restrict__ k,
                   const int* __restrict__ best_idx,
                   float* __restrict__ out_xd) {
    const size_t stride = (size_t)gridDim.x * blockDim.x;
    for (size_t o = (size_t)blockIdx.x * blockDim.x + threadIdx.x; o < (size_t)NELEM; o += stride) {
        const int t  = (int)(o & 2047);        // out layout: (N, width, T), t fastest
        const int nc = (int)(o >> 11);
        const int c  = nc & 511;
        const int nn = nc >> 9;
        const int idx = best_idx[nn * TT + t];
        out_xd[o] = k[(size_t)idx * WIDTH + c];
    }
}

// ---------------------------------------------------------------------------
extern "C" void kernel_launch(void* const* d_in, const int* in_sizes, int n_in,
                              void* d_out, int out_size, void* d_ws, size_t ws_size,
                              hipStream_t stream) {
    const float* x = (const float*)d_in[0];   // (32, 512, 2048)
    const float* k = (const float*)d_in[1];   // (2048, 512)

    // workspace layout (16B-aligned slices)
    char* ws = (char*)d_ws;
    unsigned short* kbf   = (unsigned short*)(ws);                       // 2,097,152 B
    float*          knorm = (float*)(ws + 2097152);                      //     8,192 B
    float*          bval  = (float*)(ws + 2097152 + 8192);               //   262,144 B
    int*            bidx  = (int*)  (ws + 2097152 + 8192 + 262144);      //   262,144 B
    float*          psx   = (float*)(ws + 2097152 + 8192 + 2*262144);            // 4 KB
    float*          psxx  = (float*)(ws + 2097152 + 8192 + 2*262144 + 4096);     // 4 KB
    float*          pbv   = (float*)(ws + 2097152 + 8192 + 2*262144 + 8192);     // 1 KB

    float* out_xl   = (float*)d_out;                  // 65536
    float* out_xd   = out_xl + ROWS;                  // 33554432
    float* out_tail = out_xd + (size_t)NELEM;         // 3 scalars

    prep_k_kernel<<<KBINS, 64, 0, stream>>>(k, kbf, knorm);
    stats_kernel<<<STATS_BLOCKS, 256, 0, stream>>>(x, psx, psxx);
    argmin_gemm_kernel<<<ROWS / 32, 64, 0, stream>>>(x, kbf, knorm, bval, bidx);
    reduce_bv_kernel<<<BV_BLOCKS, 256, 0, stream>>>(bval, pbv);
    finalize_kernel<<<1, 256, 0, stream>>>(psx, psxx, pbv, out_tail);
    write_xl_kernel<<<ROWS / 256, 256, 0, stream>>>(bidx, out_xl);
    gather_kernel<<<8192, 256, 0, stream>>>(k, bidx, out_xd);
}
